// KCDN_67997922230517
// MI455X (gfx1250) — matrix operand order repaired and checked by reading, not compiled, not run on hardware
//
#include <hip/hip_runtime.h>

#define NTHREADS  128
#define T_TRIPLES 128
#define DIM        64
#define N_LAYER     2
#define H_STRIDE   72   // halves per H row (64 data + 8 pad) -> 144B, 16B aligned, bank-safe

typedef __attribute__((ext_vector_type(16))) __bf16         v16bf;
typedef __attribute__((ext_vector_type(8)))  float          v8f;
typedef __attribute__((ext_vector_type(8)))  unsigned short us8;

typedef int v4i __attribute__((vector_size(16)));
typedef v4i __attribute__((address_space(1))) gv4i;   // global
typedef v4i __attribute__((address_space(3))) lv4i;   // LDS

union frag16 {
    us8   h[2];
    v16bf b;
};

#if defined(__AMDGCN__) && __has_builtin(__builtin_amdgcn_global_load_async_to_lds_b128)
#define HAVE_ASYNC_LDS 1
#else
#define HAVE_ASYNC_LDS 0
#endif

// 16-byte per-lane scattered copy global -> LDS (async on CDNA5, sync fallback).
// OFF must be a compile-time constant (immediate offset field of the instruction).
template <int OFF>
__device__ __forceinline__ void copy16_to_lds(const void* g, void* l) {
#if HAVE_ASYNC_LDS
    __builtin_amdgcn_global_load_async_to_lds_b128((gv4i*)g, (lv4i*)l, OFF, 0);
#else
    *(float4*)((char*)l + OFF) = *(const float4*)((const char*)g + OFF);
#endif
}

__device__ __forceinline__ void copy_row64f_to_lds(const void* g, void* l) {
    copy16_to_lds<0>(g, l);    copy16_to_lds<16>(g, l);
    copy16_to_lds<32>(g, l);   copy16_to_lds<48>(g, l);
    copy16_to_lds<64>(g, l);   copy16_to_lds<80>(g, l);
    copy16_to_lds<96>(g, l);   copy16_to_lds<112>(g, l);
    copy16_to_lds<128>(g, l);  copy16_to_lds<144>(g, l);
    copy16_to_lds<160>(g, l);  copy16_to_lds<176>(g, l);
    copy16_to_lds<192>(g, l);  copy16_to_lds<208>(g, l);
    copy16_to_lds<224>(g, l);  copy16_to_lds<240>(g, l);
}

__device__ __forceinline__ void wait_async_copies() {
#if HAVE_ASYNC_LDS
    asm volatile("s_wait_asynccnt 0x0" ::: "memory");
#endif
}

__device__ __forceinline__ unsigned short f2bf(float f) {
    unsigned u = __float_as_uint(f);
    u += 0x7FFFu + ((u >> 16) & 1u);          // round-to-nearest-even
    return (unsigned short)(u >> 16);
}

// pack two fp32 -> packed bf16 pair {hi16(y), hi16(x)} with one V_PERM_B32
__device__ __forceinline__ unsigned pkbf(float x, float y) {
    unsigned ux = __float_as_uint(x), uy = __float_as_uint(y);
    ux += 0x7FFFu + ((ux >> 16) & 1u);
    uy += 0x7FFFu + ((uy >> 16) & 1u);
    return __builtin_amdgcn_perm(uy, ux, 0x07060302);
}

// lane-xor butterfly step via DS_SWIZZLE_B32 group-of-32 mode:
// offset = xor_mask<<10 | or_mask<<5 | and_mask  (and_mask=0x1F, or_mask=0)
template <int XORM>
__device__ __forceinline__ float swz_xor(float v) {
    return __int_as_float(
        __builtin_amdgcn_ds_swizzle(__float_as_int(v), (XORM << 10) | 0x1F));
}

__device__ __forceinline__ float wave_max(float v) {
    v = fmaxf(v, swz_xor<16>(v));
    v = fmaxf(v, swz_xor<8>(v));
    v = fmaxf(v, swz_xor<4>(v));
    v = fmaxf(v, swz_xor<2>(v));
    v = fmaxf(v, swz_xor<1>(v));
    return v;
}

__device__ __forceinline__ float wave_sum(float v) {
    v += swz_xor<16>(v);
    v += swz_xor<8>(v);
    v += swz_xor<4>(v);
    v += swz_xor<2>(v);
    v += swz_xor<1>(v);
    return v;
}

__global__ __launch_bounds__(NTHREADS)
void kcdn_kernel(const int*   __restrict__ items,
                 const int*   __restrict__ user_h,
                 const int*   __restrict__ user_t,
                 const int*   __restrict__ item_h,
                 const int*   __restrict__ item_t,
                 const float* __restrict__ ent,
                 float*       __restrict__ out,
                 int B)
{
    __shared__ __align__(16) unsigned short Hlds[T_TRIPLES * H_STRIDE]; // bf16 heads
    __shared__ __align__(16) float          Tlds[T_TRIPLES * DIM];      // fp32 tails
    __shared__ __align__(32) unsigned short anchorB[DIM];               // bf16 anchor
    __shared__ float sims[T_TRIPLES];
    __shared__ float wexp[T_TRIPLES];
    __shared__ float partial[2][DIM];
    __shared__ float evu[2][DIM];   // evu[0]=e_v (item), evu[1]=e_u (user)
    __shared__ float red[2][4];     // cross-wave reduction slots (max / sum)

    const int b     = blockIdx.x;
    const int tid   = threadIdx.x;
    const int lane  = tid & 31;
    const int mrow  = lane & 15;          // M index inside 16x16 tile
    const int lgrp  = (lane >> 4) & 1;    // lane half-group
    const int kbA   = lgrp * 8;           // A: K base {0..7,16..23} vs {8..15,24..31}
    const int kbB   = lgrp * 16;          // unused: B fragments also use kbA (same k order as A)
    const int wave  = tid >> 5;
    const int p     = tid >> 6;           // 0/1 split over triples
    const int d     = tid & 63;           // dim index

    for (int side = 0; side < 2; ++side) {
        const int* hIdx = side ? user_h : item_h;
        const int* tIdx = side ? user_t : item_t;

        // ---------------- anchor ----------------
        if (side == 0) {
            if (tid < DIM) {
                float v = ent[(size_t)items[b] * DIM + tid];
                anchorB[tid] = f2bf(v);
                evu[0][tid]  = v;
            }
        } else {
            // user_0 = mean_t ent[user_h[0,b,t]]
            const int* h0 = user_h + (size_t)b * T_TRIPLES;
            float acc = 0.f;
            for (int t = p * 64; t < p * 64 + 64; ++t)
                acc += ent[(size_t)h0[t] * DIM + d];
            partial[p][d] = acc;
            __syncthreads();
            if (tid < DIM) {
                float v = (partial[0][d] + partial[1][d]) * (1.0f / 128.0f);
                anchorB[d] = f2bf(v);
                evu[1][d]  = v;
            }
        }
        __syncthreads();

        // B fragments: anchor replicated across all 16 columns, one per k-step
        v16bf bfrag[2];
        #pragma unroll
        for (int ks = 0; ks < 2; ++ks) {
            frag16 f;
            const int o = ks * 32 + kbA;
            f.h[0] = *(const us8*)&anchorB[o];
            f.h[1] = *(const us8*)&anchorB[o + 16];
            bfrag[ks] = f.b;
        }

        for (int layer = 0; layer < N_LAYER; ++layer) {
            // ------------- gather: T via async LDS copies, H via load+bf16-convert ----
            {
                const size_t base = ((size_t)layer * B + b) * T_TRIPLES + tid;
                const int hr = hIdx[base];
                const int tr = tIdx[base];

                // T row: 16B-per-lane scattered async copies straight into LDS
                copy_row64f_to_lds(ent + (size_t)tr * DIM, &Tlds[tid * DIM]);

                // H row: load fp32, pack to bf16 (V_PERM_B32), store to padded LDS
                const float4* hp = (const float4*)(ent + (size_t)hr * DIM);
                unsigned short* hd = &Hlds[tid * H_STRIDE];
                #pragma unroll
                for (int q = 0; q < DIM / 4; ++q) {
                    float4 hv = hp[q];
                    uint2  pk;
                    pk.x = pkbf(hv.x, hv.y);
                    pk.y = pkbf(hv.z, hv.w);
                    *(uint2*)&hd[q * 4] = pk;
                }
            }
            wait_async_copies();
            __syncthreads();

            // ------------- sims[t] = H . anchor via WMMA (bf16 -> f32) -------------
            #pragma unroll
            for (int rep = 0; rep < 2; ++rep) {
                const int t0 = (wave + rep * 4) * 16;
                v8f c = {};
                #pragma unroll
                for (int ks = 0; ks < 2; ++ks) {
                    frag16 fa;
                    const int off = (t0 + mrow) * H_STRIDE + ks * 32 + kbA;
                    fa.h[0] = *(const us8*)&Hlds[off];
                    fa.h[1] = *(const us8*)&Hlds[off + 16];
                    c = __builtin_amdgcn_wmma_f32_16x16x32_bf16(
                            false, fa.b, false, bfrag[ks],
                            (short)0, c, false, false);
                }
                // D column N=0: lane 0 holds M=r, lane 16 holds M=8+r
                if (mrow == 0) {
                    #pragma unroll
                    for (int r = 0; r < 8; ++r)
                        sims[t0 + lgrp * 8 + r] = c[r];
                }
            }
            __syncthreads();

            // ------------- softmax over 128 triples (butterfly reductions) -------------
            const float sv = sims[tid];
            float wm = wave_max(sv);
            if (lane == 0) red[0][wave] = wm;
            __syncthreads();
            const float mx = fmaxf(fmaxf(red[0][0], red[0][1]),
                                   fmaxf(red[0][2], red[0][3]));
            const float e = __expf(sv - mx);
            wexp[tid] = e;
            float ws = wave_sum(e);
            if (lane == 0) red[1][wave] = ws;
            __syncthreads();   // wexp + partial sums visible
            const float rdenom = 1.0f /
                (red[1][0] + red[1][1] + red[1][2] + red[1][3]);

            // ------------- weighted tail sum (fp32, conflict-free LDS) -------------
            float acc = 0.f;
            for (int t = p * 64; t < p * 64 + 64; ++t)
                acc += wexp[t] * Tlds[t * DIM + d];
            partial[p][d] = acc;
            __syncthreads();
            if (tid < DIM)
                evu[side][d] += (partial[0][d] + partial[1][d]) * rdenom;
            __syncthreads();
        }
    }

    // ---------------- score = sigmoid(e_v . e_u) ----------------
    if (tid < DIM) {
        float s = wave_sum(evu[0][d] * evu[1][d]);
        if (lane == 0) red[0][wave] = s;
    }
    __syncthreads();
    if (tid == 0) {
        float s = red[0][0] + red[0][1];
        out[b] = 1.0f / (1.0f + __expf(-s));
    }
}

extern "C" void kernel_launch(void* const* d_in, const int* in_sizes, int n_in,
                              void* d_out, int out_size, void* d_ws, size_t ws_size,
                              hipStream_t stream) {
    // setup_inputs order:
    // 0 items, 1 user_h, 2 user_r, 3 user_t, 4 item_h, 5 item_r, 6 item_t,
    // 7 entity_emb, 8 relation_emb  (r / relation_emb unused by the reference)
    const int*   items  = (const int*)  d_in[0];
    const int*   user_h = (const int*)  d_in[1];
    const int*   user_t = (const int*)  d_in[3];
    const int*   item_h = (const int*)  d_in[4];
    const int*   item_t = (const int*)  d_in[6];
    const float* ent    = (const float*)d_in[7];
    float* out = (float*)d_out;

    const int B = in_sizes[0];   // 4096
    kcdn_kernel<<<B, NTHREADS, 0, stream>>>(items, user_h, user_t,
                                            item_h, item_t, ent, out, B);
}
